// CrossAttention_14070312861870
// MI455X (gfx1250) — matrix operand order repaired and checked by reading, not compiled, not run on hardware
//
#include <hip/hip_runtime.h>
#include <hip/hip_bf16.h>

// ---------------------------------------------------------------------------
// CDNA5 (gfx1250) cross-attention: wave32 WMMA bf16 + TDM (tensor_load_to_lds)
// double-buffered LDS staging shared by all 4 waves of a block.
// Pipeline:
//   1. transpose/convert fp32 inputs & weights to bf16 (in_aT, in_bT, WaT, WbT)
//   2. proj:       mapped = bf16( X @ W + bias )              [WMMA]
//   3. rowsum:     S_row = sum_m exp(A@B^T), S_col = sum_l    [WMMA + TDM]
//   4. attn_out:   O = (exp(A@B^T) * invSum) @ V              [WMMA x2 + TDM]
//   5. pad copy into the middle third of the output
// ---------------------------------------------------------------------------

typedef __bf16 bf16_t;
typedef __attribute__((ext_vector_type(16))) __bf16 v16bf;
typedef __attribute__((ext_vector_type(8)))  __bf16 v8bf;
typedef __attribute__((ext_vector_type(8)))  float  v8f;
typedef unsigned int u32x4 __attribute__((ext_vector_type(4)));
typedef int          i32x4 __attribute__((ext_vector_type(4)));
typedef int          i32x8 __attribute__((ext_vector_type(8)));

static constexpr int Bn = 4;
static constexpr int Ln = 2048;
static constexpr int Dn = 256;            // Da == Db
static constexpr int Hn = 256;
static constexpr int OUT_STRIDE = 3 * Dn; // 768
static constexpr float SHIFT = 16.0f;     // cancels in softmax, avoids exp overflow
static constexpr int NCHUNK = Ln / 32;    // 64 chunks of 32 m-rows

// ---- TDM: 2D bf16 tile DMA global -> LDS (D# per CDNA5 ISA §8.3/8.4) ------
// tile_w elems contiguous per row, tile_h rows, row pitch row_stride elems.
__device__ __forceinline__ void tdm_load_2d(const void* gptr, unsigned lds_off,
                                            int tile_w, int tile_h,
                                            int row_stride_elems) {
  unsigned long long ga = (unsigned long long)(uintptr_t)gptr;
  u32x4 g0;
  g0.x = 1u;                                            // count=1, user mode
  g0.y = lds_off;                                       // lds_addr (bytes)
  g0.z = (unsigned)ga;                                  // global_addr[31:0]
  g0.w = (unsigned)((ga >> 32) & 0x01FFFFFFu) | (2u << 30);  // addr[56:32]|type=2
  i32x8 g1;
  g1[0] = (1 << 16);                                    // data_size=1 (2 bytes)
  g1[1] = (tile_w & 0xffff) << 16;                      // tensor_dim0[15:0]
  g1[2] = ((unsigned)tile_w >> 16) | ((tile_h & 0xffff) << 16); // dim0 hi|dim1 lo
  g1[3] = ((unsigned)tile_h >> 16) | ((tile_w & 0xffff) << 16); // dim1 hi|tile_dim0
  g1[4] = (tile_h & 0xffff);                            // tile_dim1 (tile_dim2=0)
  g1[5] = row_stride_elems;                             // tensor_dim0_stride lo32
  g1[6] = 0;
  g1[7] = 0;
  i32x4 zz = {0, 0, 0, 0};
#if defined(__clang_major__) && (__clang_major__ >= 23)
  i32x8 z8 = {0, 0, 0, 0, 0, 0, 0, 0};
  __builtin_amdgcn_tensor_load_to_lds(g0, g1, zz, zz, z8, 0);
#else
  __builtin_amdgcn_tensor_load_to_lds(g0, g1, zz, zz, 0);
#endif
}

// ---- fragment loaders -----------------------------------------------------
// A matrix 16x32 bf16 (MxK), row-major source, leading dim `ld`.
__device__ __forceinline__ v16bf load_frag_a(const bf16_t* __restrict__ src,
                                             int ld, int lane) {
  const int row  = lane & 15;
  const int colb = ((lane >> 4) & 1) * 8;
  const bf16_t* p = src + (size_t)row * ld + colb;
  v8bf lo = *(const v8bf*)(p);
  v8bf hi = *(const v8bf*)(p + 16);
  v16bf a;
#pragma unroll
  for (int i = 0; i < 8; ++i) { a[i] = lo[i]; a[i + 8] = hi[i]; }
  return a;
}

// Same shape, fp32 source with on-the-fly convert.
__device__ __forceinline__ v16bf load_frag_a_f32(const float* __restrict__ src,
                                                 int ld, int lane) {
  const int row  = lane & 15;
  const int colb = ((lane >> 4) & 1) * 8;
  const float* p = src + (size_t)row * ld + colb;
  v16bf a;
#pragma unroll
  for (int i = 0; i < 8; ++i) {
    a[i]     = (bf16_t)p[i];
    a[i + 8] = (bf16_t)p[i + 16];
  }
  return a;
}

// B matrix 32x16 bf16 (KxN), "N-major" source: element (k,n) = srcT[n*ld + k].
// Works for both global and LDS pointers.
__device__ __forceinline__ v16bf load_frag_b(const bf16_t* srcT, int ld,
                                             int lane) {
  const int row = lane & 15;
  const int kb  = ((lane >> 4) & 1) * 8;
  const bf16_t* p = srcT + (size_t)row * ld + kb;
  v8bf lo = *(const v8bf*)(p);
  v8bf hi = *(const v8bf*)(p + 16);
  v16bf b;
#pragma unroll
  for (int i = 0; i < 8; ++i) { b[i] = lo[i]; b[i + 8] = hi[i]; }
  return b;
}

// ---- 1. transpose + fp32->bf16 convert ------------------------------------
__global__ void transpose_bf16_kernel(const float* __restrict__ src,
                                      bf16_t* __restrict__ dst,
                                      int rows, int cols, int batches) {
  int idx = blockIdx.x * blockDim.x + threadIdx.x;
  int total = batches * rows * cols;
  if (idx >= total) return;
  int b   = idx / (rows * cols);
  int rem = idx - b * rows * cols;
  int r   = rem / cols;
  int c   = rem - r * cols;
  dst[(size_t)b * rows * cols + (size_t)c * rows + r] = (bf16_t)src[idx];
}

// ---- 2. projection: mapped = bf16(X @ W + bias) ---------------------------
__global__ __launch_bounds__(128) void proj_kernel(
    const float* __restrict__ X, const bf16_t* __restrict__ WT,
    const float* __restrict__ bias, bf16_t* __restrict__ M) {
  const int lane  = threadIdx.x & 31;
  const int wv    = threadIdx.x >> 5;
  const int strip = blockIdx.x * 4 + wv;
  const int b  = strip >> 7;
  const int lt = strip & 127;

  const float* Xb = X + (size_t)(b * Ln + lt * 16) * Dn;
  v16bf afrag[8];
#pragma unroll
  for (int kc = 0; kc < 8; ++kc)
    afrag[kc] = load_frag_a_f32(Xb + kc * 32, Dn, lane);

  const int col  = lane & 15;
  const int half = (lane >> 4) & 1;
  bf16_t* Mb = M + (size_t)(b * Ln + lt * 16) * Hn;

  for (int ht = 0; ht < 16; ++ht) {
    v8f acc = {};
#pragma unroll
    for (int kc = 0; kc < 8; ++kc) {
      v16bf bfrag = load_frag_b(WT + (size_t)ht * 16 * Dn + kc * 32, Dn, lane);
      acc = __builtin_amdgcn_wmma_f32_16x16x32_bf16(false, afrag[kc], false,
                                                    bfrag, (short)0, acc,
                                                    false, false);
    }
    const float bv = bias[ht * 16 + col];
#pragma unroll
    for (int r = 0; r < 8; ++r)
      Mb[(size_t)(r + 8 * half) * Hn + ht * 16 + col] = (bf16_t)(acc[r] + bv);
  }
}

// ---- 3. exp-row-sums of X @ Y^T (Y staged via TDM, shared by 4 waves) -----
__global__ __launch_bounds__(128) void score_rowsum_kernel(
    const bf16_t* __restrict__ mapX, const bf16_t* __restrict__ mapY,
    float* __restrict__ Ssum) {
  __shared__ alignas(16) struct { bf16_t y[2][32 * Hn]; } sm;  // 32 KB
  const int lane = threadIdx.x & 31;
  const int wv   = threadIdx.x >> 5;
  const int b  = blockIdx.x >> 5;              // 32 blocks per batch
  const int lt = (blockIdx.x & 31) * 4 + wv;   // this wave's 16-row strip

  const bf16_t* Xb = mapX + (size_t)(b * Ln + lt * 16) * Hn;
  const bf16_t* Yb = mapY + (size_t)b * Ln * Hn;
  const unsigned yoff[2] = {
      (unsigned)((const char*)&sm.y[0][0] - (const char*)&sm),
      (unsigned)((const char*)&sm.y[1][0] - (const char*)&sm)};

  v16bf afrag[8];
#pragma unroll
  for (int kc = 0; kc < 8; ++kc)
    afrag[kc] = load_frag_a(Xb + kc * 32, Hn, lane);

  if (wv == 0)   // prefetch chunk 0
    tdm_load_2d(Yb, yoff[0], Hn, 32, Hn);

  float rp[8] = {};
  for (int mc = 0; mc < NCHUNK; ++mc) {
    if (wv == 0) {
      if (mc + 1 < NCHUNK) {
        tdm_load_2d(Yb + (size_t)(mc + 1) * 32 * Hn, yoff[(mc + 1) & 1],
                    Hn, 32, Hn);
        __builtin_amdgcn_s_wait_tensorcnt(1);   // chunk mc complete
      } else {
        __builtin_amdgcn_s_wait_tensorcnt(0);
      }
    }
    __syncthreads();
    const bf16_t* yb = sm.y[mc & 1];
#pragma unroll
    for (int sub = 0; sub < 2; ++sub) {
      v8f acc = {};
#pragma unroll
      for (int kc = 0; kc < 8; ++kc) {
        v16bf bfrag = load_frag_b(yb + (size_t)sub * 16 * Hn + kc * 32, Hn, lane);
        acc = __builtin_amdgcn_wmma_f32_16x16x32_bf16(false, afrag[kc], false,
                                                      bfrag, (short)0, acc,
                                                      false, false);
      }
#pragma unroll
      for (int r = 0; r < 8; ++r) rp[r] += __expf(acc[r] - SHIFT);
    }
    __syncthreads();
  }
  // reduce across the 16 lanes of each half-wave (xor masks stay within half)
#pragma unroll
  for (int m = 1; m <= 8; m <<= 1)
#pragma unroll
    for (int r = 0; r < 8; ++r) rp[r] += __shfl_xor(rp[r], m, 32);

  if ((lane & 15) == 0) {
    const int half = lane >> 4;
#pragma unroll
    for (int r = 0; r < 8; ++r)
      Ssum[b * Ln + lt * 16 + r + 8 * half] = rp[r];
  }
}

// ---- 4. attention output: O = (exp(X@Y^T)*invSum) @ V ---------------------
__global__ __launch_bounds__(128) void attn_out_kernel(
    const bf16_t* __restrict__ mapX, const bf16_t* __restrict__ mapY,
    const bf16_t* __restrict__ VT, const float* __restrict__ sumOther,
    float* __restrict__ outp, int outOffset) {
  __shared__ alignas(16) struct {
    bf16_t y[2][32 * Hn];   // 32 KB : mapped_Y chunk  [32 m][256 h]
    bf16_t v[2][Dn * 32];   // 32 KB : V^T chunk       [256 d][32 m]
    bf16_t p[4][16 * 32];   //  4 KB : per-wave P tile
  } sm;
  const int lane = threadIdx.x & 31;
  const int wv   = threadIdx.x >> 5;
  const int b  = blockIdx.x >> 5;
  const int lt = (blockIdx.x & 31) * 4 + wv;

  const bf16_t* Xb = mapX + (size_t)(b * Ln + lt * 16) * Hn;
  const bf16_t* Yb = mapY + (size_t)b * Ln * Hn;
  const bf16_t* Vb = VT   + (size_t)b * Dn * Ln;
  const float*  Sb = sumOther + b * Ln;
  const unsigned yoff[2] = {
      (unsigned)((const char*)&sm.y[0][0] - (const char*)&sm),
      (unsigned)((const char*)&sm.y[1][0] - (const char*)&sm)};
  const unsigned voff[2] = {
      (unsigned)((const char*)&sm.v[0][0] - (const char*)&sm),
      (unsigned)((const char*)&sm.v[1][0] - (const char*)&sm)};

  v16bf afrag[8];
#pragma unroll
  for (int kc = 0; kc < 8; ++kc)
    afrag[kc] = load_frag_a(Xb + kc * 32, Hn, lane);

  v8f O[16];
#pragma unroll
  for (int t = 0; t < 16; ++t) O[t] = (v8f){};

  const int col  = lane & 15;
  const int half = (lane >> 4) & 1;
  bf16_t* pb = &sm.p[wv][0];

  if (wv == 0) {  // prefetch chunk 0 (Y and V)
    tdm_load_2d(Yb, yoff[0], Hn, 32, Hn);
    tdm_load_2d(Vb, voff[0], 32, Dn, Ln);
  }

  for (int mc = 0; mc < NCHUNK; ++mc) {
    if (wv == 0) {
      if (mc + 1 < NCHUNK) {
        tdm_load_2d(Yb + (size_t)(mc + 1) * 32 * Hn, yoff[(mc + 1) & 1],
                    Hn, 32, Hn);
        tdm_load_2d(Vb + (size_t)(mc + 1) * 32, voff[(mc + 1) & 1],
                    32, Dn, Ln);
        __builtin_amdgcn_s_wait_tensorcnt(2);   // chunk mc pair complete
      } else {
        __builtin_amdgcn_s_wait_tensorcnt(0);
      }
    }
    __syncthreads();
    const bf16_t* yb = sm.y[mc & 1];
    const bf16_t* vb = sm.v[mc & 1];
#pragma unroll
    for (int sub = 0; sub < 2; ++sub) {
      v8f acc = {};
#pragma unroll
      for (int kc = 0; kc < 8; ++kc) {
        v16bf bfrag = load_frag_b(yb + (size_t)sub * 16 * Hn + kc * 32, Hn, lane);
        acc = __builtin_amdgcn_wmma_f32_16x16x32_bf16(false, afrag[kc], false,
                                                      bfrag, (short)0, acc,
                                                      false, false);
      }
      const float inv = 1.0f / Sb[mc * 32 + sub * 16 + col];
      // C/D layout: lane holds (M = r + 8*half, N = col) -> scatter into LDS
      // as row-major 16(l) x 32(m) bf16 P tile.
#pragma unroll
      for (int r = 0; r < 8; ++r) {
        float w = __expf(acc[r] - SHIFT) * inv;
        pb[(r + 8 * half) * 32 + sub * 16 + col] = (bf16_t)w;
      }
    }
    // reload P tile in A-fragment layout (same-wave LDS RAW, dscnt enforced)
    v16bf pfrag;
    {
      const bf16_t* p = pb + (lane & 15) * 32 + half * 8;
      v8bf lo = *(const v8bf*)(p);
      v8bf hi = *(const v8bf*)(p + 16);
#pragma unroll
      for (int i = 0; i < 8; ++i) { pfrag[i] = lo[i]; pfrag[i + 8] = hi[i]; }
    }
#pragma unroll
    for (int dt = 0; dt < 16; ++dt) {
      v16bf vfrag = load_frag_b(vb + (size_t)dt * 16 * 32, 32, lane);
      O[dt] = __builtin_amdgcn_wmma_f32_16x16x32_bf16(false, pfrag, false,
                                                      vfrag, (short)0, O[dt],
                                                      false, false);
    }
    __syncthreads();
  }

  float* ob = outp + (size_t)(b * Ln + lt * 16) * OUT_STRIDE + outOffset;
#pragma unroll
  for (int dt = 0; dt < 16; ++dt)
#pragma unroll
    for (int r = 0; r < 8; ++r)
      ob[(size_t)(r + 8 * half) * OUT_STRIDE + dt * 16 + col] = O[dt][r];
}

// ---- 5. padding copy into middle third ------------------------------------
__global__ void pad_copy_kernel(const float* __restrict__ pad,
                                float* __restrict__ outp) {
  int idx = blockIdx.x * blockDim.x + threadIdx.x;
  if (idx >= Bn * Ln * Dn) return;
  int d  = idx % Dn;
  int bl = idx / Dn;
  outp[(size_t)bl * OUT_STRIDE + Dn + d] = pad[idx];
}

// ---------------------------------------------------------------------------
extern "C" void kernel_launch(void* const* d_in, const int* in_sizes, int n_in,
                              void* d_out, int out_size, void* d_ws, size_t ws_size,
                              hipStream_t stream) {
  const float* in_a = (const float*)d_in[0];
  const float* in_b = (const float*)d_in[1];
  const float* Wa   = (const float*)d_in[2];
  const float* ba   = (const float*)d_in[3];
  const float* Wb   = (const float*)d_in[4];
  const float* bb   = (const float*)d_in[5];
  const float* pad  = (const float*)d_in[6];
  float* outp = (float*)d_out;

  char* ws = (char*)d_ws;
  size_t off = 0;
  bf16_t* mapA = (bf16_t*)(ws + off); off += (size_t)Bn * Ln * Hn * 2;
  bf16_t* mapB = (bf16_t*)(ws + off); off += (size_t)Bn * Ln * Hn * 2;
  bf16_t* inAT = (bf16_t*)(ws + off); off += (size_t)Bn * Dn * Ln * 2;
  bf16_t* inBT = (bf16_t*)(ws + off); off += (size_t)Bn * Dn * Ln * 2;
  bf16_t* WaT  = (bf16_t*)(ws + off); off += (size_t)Dn * Hn * 2;
  bf16_t* WbT  = (bf16_t*)(ws + off); off += (size_t)Dn * Hn * 2;
  float*  Srow = (float*)(ws + off);  off += (size_t)Bn * Ln * 4;
  float*  Scol = (float*)(ws + off);  off += (size_t)Bn * Ln * 4;
  (void)ws_size; (void)in_sizes; (void)n_in; (void)out_size;

  const int strips = Bn * (Ln / 16);      // 512 strips, 4 waves / block
  dim3 blk(128), grd(strips / 4);

  const int totIn = Bn * Ln * Dn;
  transpose_bf16_kernel<<<(totIn + 255) / 256, 256, 0, stream>>>(in_a, inAT, Ln, Dn, Bn);
  transpose_bf16_kernel<<<(totIn + 255) / 256, 256, 0, stream>>>(in_b, inBT, Ln, Dn, Bn);
  transpose_bf16_kernel<<<(Dn * Hn + 255) / 256, 256, 0, stream>>>(Wa, WaT, Dn, Hn, 1);
  transpose_bf16_kernel<<<(Dn * Hn + 255) / 256, 256, 0, stream>>>(Wb, WbT, Dn, Hn, 1);

  proj_kernel<<<grd, blk, 0, stream>>>(in_a, WaT, ba, mapA);
  proj_kernel<<<grd, blk, 0, stream>>>(in_b, WbT, bb, mapB);

  score_rowsum_kernel<<<grd, blk, 0, stream>>>(mapA, mapB, Srow);  // S_row[l]
  score_rowsum_kernel<<<grd, blk, 0, stream>>>(mapB, mapA, Scol);  // S_col[m]

  // output_a = (exp(S)/S_col) @ input_b  -> cols [0,256)
  attn_out_kernel<<<grd, blk, 0, stream>>>(mapA, mapB, inBT, Scol, outp, 0);
  // output_b = (exp(S^T)/S_row) @ input_a -> cols [512,768)
  attn_out_kernel<<<grd, blk, 0, stream>>>(mapB, mapA, inAT, Srow, outp, 2 * Dn);

  pad_copy_kernel<<<(totIn + 255) / 256, 256, 0, stream>>>(pad, outp);
}
